// Pre_FairADG_6296422056682
// MI455X (gfx1250) — matrix operand order repaired and checked by reading, not compiled, not run on hardware
//
#include <hip/hip_runtime.h>
#include <hip/hip_bf16.h>
#include <math.h>

typedef __attribute__((ext_vector_type(16))) _Float16 v16h;
typedef __attribute__((ext_vector_type(8)))  float    v8f;

#define NFEAT 128
#define KCH   4
#define PCD   32
// B matrix is 128 x 144: cols 0..127 = Weff (Wlin@Wconv folded), 128..131 = aW1[:F],
// 132..135 = aW1[F:], 136..143 = zero pad.  4 k-tiles x 9 n-tiles of WMMA fragments.
#define NT    9
#define KT    4
#define NFRAG (KT * NT)   // 36 fragments, each 32 lanes x 16 halves

// ---------------- setup: fold Wconv into Wlin, build swizzled f16 B, beff -------------
__global__ void k_setup(const float* __restrict__ Wlin, const float* __restrict__ Wconv,
                        const float* __restrict__ aW1, const float* __restrict__ blin,
                        _Float16* __restrict__ Bswz, float* __restrict__ beff) {
  int tid = threadIdx.x;
  // beff[col] = (blin @ Wconv) per channel (blin is zeros in reference, keep general)
  for (int col = tid; col < 128; col += 256) {
    int k = col >> 5, p = col & 31;
    float s = 0.f;
    for (int q = 0; q < 32; ++q) s += blin[k * 32 + q] * Wconv[(k * 32 + q) * 32 + p];
    beff[col] = s;
  }
  // Swizzled B fragments, element e of lane L of frag(kt,nt):
  //   K = kt*32 + (e>=8 ? 16 : 0) + (L>=16 ? 8 : 0) + (e&7) ,  col = nt*16 + (L&15)   (ISA f16 B 32x16 layout)
  for (int idx = tid; idx < NFRAG * 32 * 16; idx += 256) {
    int e    = idx & 15;
    int lane = (idx >> 4) & 31;
    int frag = idx >> 9;
    int kt = frag / NT, nt = frag % NT;
    int krow = kt * 32 + ((e & 8) << 1) + ((lane >= 16) ? 8 : 0) + (e & 7);   // 0..127
    int col  = nt * 16 + (lane & 15);                   // 0..143
    float v;
    if (col < 128) {                 // Weff = Wlin[k] @ Wconv[k]
      int k = col >> 5, p = col & 31;
      float s = 0.f;
      for (int q = 0; q < 32; ++q)
        s += Wlin[((size_t)k * 128 + krow) * 32 + q] * Wconv[(k * 32 + q) * 32 + p];
      v = s;
    } else if (col < 132) {          // p-projection weights: aW1[:128]
      v = aW1[krow * 4 + (col - 128)];
    } else if (col < 136) {          // q-projection weights: aW1[128:]
      v = aW1[(128 + krow) * 4 + (col - 132)];
    } else {
      v = 0.f;
    }
    Bswz[idx] = (_Float16)v;
  }
}

// ---------------- WMMA GEMM: c[N,128] = x @ Weff + beff ;  pq[N,8] = x @ [aW1lo|aW1hi] ---
__global__ void __launch_bounds__(256) k_gemm(const float* __restrict__ x,
                                              const _Float16* __restrict__ Bswz,
                                              const float* __restrict__ beff,
                                              float* __restrict__ c,
                                              float* __restrict__ pq, int N) {
  __shared__ v16h Bsh[NFRAG * 32];               // 36 KB
  int tid = threadIdx.x;
  {
    const float4* src = (const float4*)Bswz;     // 36*32*32B = 2304 x 16B
    float4* dst = (float4*)Bsh;
    for (int i = tid; i < NFRAG * 32 * 2; i += 256) dst[i] = src[i];
  }
  __syncthreads();

  int wave = tid >> 5, lane = tid & 31;
  int r0 = (blockIdx.x * 8 + wave) * 16;
  if (r0 >= N) return;
  int halfhi = (lane >= 16) ? 1 : 0;
  int lm = lane & 15;

  v8f acc[NT];
#pragma unroll
  for (int nt = 0; nt < NT; ++nt) {
    float b = (nt < 8) ? beff[nt * 16 + lm] : 0.f;
#pragma unroll
    for (int r = 0; r < 8; ++r) acc[nt][r] = b;
  }

  int m = r0 + lm; if (m >= N) m = N - 1;        // clamp tail reads
  const float* xrow = x + (size_t)m * NFEAT;

#pragma unroll
  for (int kt = 0; kt < KT; ++kt) {
    // A-fragment, ISA f16 16x32 layout: lane holds K = kb..kb+7 and kb+16..kb+23
    int kb = kt * 32 + halfhi * 8;
    float4 a0 = *(const float4*)(xrow + kb);
    float4 a1 = *(const float4*)(xrow + kb + 4);
    float4 a2 = *(const float4*)(xrow + kb + 16);
    float4 a3 = *(const float4*)(xrow + kb + 20);
    v16h A;
    A[0]=(_Float16)a0.x; A[1]=(_Float16)a0.y; A[2]=(_Float16)a0.z; A[3]=(_Float16)a0.w;
    A[4]=(_Float16)a1.x; A[5]=(_Float16)a1.y; A[6]=(_Float16)a1.z; A[7]=(_Float16)a1.w;
    A[8]=(_Float16)a2.x; A[9]=(_Float16)a2.y; A[10]=(_Float16)a2.z; A[11]=(_Float16)a2.w;
    A[12]=(_Float16)a3.x; A[13]=(_Float16)a3.y; A[14]=(_Float16)a3.z; A[15]=(_Float16)a3.w;
#pragma unroll
    for (int nt = 0; nt < NT; ++nt) {
      v16h B = Bsh[(kt * NT + nt) * 32 + lane];
      acc[nt] = __builtin_amdgcn_wmma_f32_16x16x32_f16(
          false, A, false, B, (short)0, acc[nt], false, false);
    }
  }

  // C/D f32 16x16 layout: VGPR r -> row r (lanes 0-15) / row 8+r (lanes 16-31), col = lane&15
#pragma unroll
  for (int nt = 0; nt < 8; ++nt) {
#pragma unroll
    for (int r = 0; r < 8; ++r) {
      int row = r0 + r + halfhi * 8;
      if (row < N) c[(size_t)row * 128 + nt * 16 + lm] = acc[nt][r];
    }
  }
  if (lm < 8) {   // 9th tile: cols 128..135 -> pq[N][8] (p = 0..3, q = 4..7)
#pragma unroll
    for (int r = 0; r < 8; ++r) {
      int row = r0 + r + halfhi * 8;
      if (row < N) pq[(size_t)row * 8 + lm] = acc[8][r];
    }
  }
}

// ---------------- CSR build ------------------------------------------------------------
__global__ void k_zero(int* __restrict__ a, int n) {
  int i = blockIdx.x * 256 + threadIdx.x;
  if (i < n) a[i] = 0;
}

__global__ void k_hist(const int* __restrict__ row, int E, int* __restrict__ deg) {
  int i = blockIdx.x * 256 + threadIdx.x;
  if (i < E) atomicAdd(&deg[row[i]], 1);
}

__global__ void __launch_bounds__(1024) k_scan(const int* __restrict__ deg,
                                               int* __restrict__ starts,
                                               int* __restrict__ cursors, int N) {
  __shared__ int part[1024];
  __shared__ int total;
  int tid = threadIdx.x;
  int chunk = (N + 1023) / 1024;
  int lo = tid * chunk;
  int hi = lo + chunk; if (hi > N) hi = N;
  int s = 0;
  for (int i = lo; i < hi; ++i) s += deg[i];
  part[tid] = s;
  __syncthreads();
  if (tid == 0) {
    int run = 0;
    for (int i = 0; i < 1024; ++i) { int t = part[i]; part[i] = run; run += t; }
    total = run;
  }
  __syncthreads();
  int run = part[tid];
  for (int i = lo; i < hi; ++i) {
    starts[i] = run; cursors[i] = run;
    run += deg[i];
  }
  if (tid == 0) starts[N] = total;
}

__global__ void k_fill(const int* __restrict__ row, int E,
                       int* __restrict__ cursors, int* __restrict__ eid) {
  int i = blockIdx.x * 256 + threadIdx.x;
  if (i < E) { int p = atomicAdd(&cursors[row[i]], 1); eid[p] = i; }
}

// ---------------- aggregation + softmax edge weights + normalize + classifier ----------
// One wave per node. Lane owns cols [4*lane, 4*lane+3] (one channel per 8-lane group).
__global__ void __launch_bounds__(256) k_agg(const float* __restrict__ c,
                                             const float* __restrict__ pq,
                                             const int* __restrict__ starts,
                                             const int* __restrict__ eid,
                                             const int* __restrict__ ecol,
                                             const float* __restrict__ ab1,
                                             const float* __restrict__ aW2,
                                             const float* __restrict__ ab2,
                                             const float* __restrict__ bch,
                                             const float* __restrict__ Wcls,
                                             const float* __restrict__ bcls,
                                             float* __restrict__ h,
                                             float* __restrict__ logits, int N) {
  int wave = threadIdx.x >> 5, lane = threadIdx.x & 31;
  int node = blockIdx.x * 8 + wave;
  if (node >= N) return;

  int s  = starts[node];
  int e1 = starts[node + 1];
  float4 qv = *(const float4*)(pq + (size_t)node * 8 + 4);   // q at the row node
  float b1x = ab1[0], b1y = ab1[1], b1z = ab1[2], b1w = ab1[3];
  float b2[4]; b2[0] = ab2[0]; b2[1] = ab2[1]; b2[2] = ab2[2]; b2[3] = ab2[3];
  float W2[16];
#pragma unroll
  for (int j = 0; j < 16; ++j) W2[j] = aW2[j];
  int ch = lane >> 3;

  float ax = 0.f, ay = 0.f, az = 0.f, aw = 0.f;
  for (int i = s; i < e1; ++i) {
    int e   = eid[i];
    int col = ecol[e];
    float4 pv = *(const float4*)(pq + (size_t)col * 8);      // p at the col node
    float h1[4];
    h1[0] = pv.x + qv.x + b1x; h1[1] = pv.y + qv.y + b1y;
    h1[2] = pv.z + qv.z + b1z; h1[3] = pv.w + qv.w + b1w;
    float t[4];
#pragma unroll
    for (int j = 0; j < 4; ++j)
      t[j] = b2[j] + h1[0]*W2[0*4+j] + h1[1]*W2[1*4+j] + h1[2]*W2[2*4+j] + h1[3]*W2[3*4+j];
    float mx = fmaxf(fmaxf(t[0], t[1]), fmaxf(t[2], t[3]));
    float ex0 = __expf(t[0]-mx), ex1 = __expf(t[1]-mx), ex2 = __expf(t[2]-mx), ex3 = __expf(t[3]-mx);
    float inv = 1.f / (ex0 + ex1 + ex2 + ex3);
    float wgt = ((ch == 0) ? ex0 : (ch == 1) ? ex1 : (ch == 2) ? ex2 : ex3) * inv;

    float4 cv = *(const float4*)(c + (size_t)col * 128 + lane * 4);
    ax += wgt * cv.x; ay += wgt * cv.y; az += wgt * cv.z; aw += wgt * cv.w;
  }

  float4 bv = *(const float4*)(bch + lane * 4);
  ax += bv.x; ay += bv.y; az += bv.z; aw += bv.w;

  // per-channel L2 norm over 32 cols = 8 lanes (xor within 8-lane group)
  float s2 = ax*ax + ay*ay + az*az + aw*aw;
  s2 += __shfl_xor(s2, 1, 32);
  s2 += __shfl_xor(s2, 2, 32);
  s2 += __shfl_xor(s2, 4, 32);
  float invn = 1.f / fmaxf(sqrtf(s2), 1e-12f);
  float4 hv; hv.x = ax * invn; hv.y = ay * invn; hv.z = az * invn; hv.w = aw * invn;
  *(float4*)(h + (size_t)node * 128 + lane * 4) = hv;

  float4 wv = *(const float4*)(Wcls + lane * 4);
  float pl = hv.x*wv.x + hv.y*wv.y + hv.z*wv.z + hv.w*wv.w;
  pl += __shfl_xor(pl, 1, 32);
  pl += __shfl_xor(pl, 2, 32);
  pl += __shfl_xor(pl, 4, 32);
  pl += __shfl_xor(pl, 8, 32);
  pl += __shfl_xor(pl, 16, 32);
  if (lane == 0) logits[node] = pl + bcls[0];
}

// ---------------- host launcher --------------------------------------------------------
extern "C" void kernel_launch(void* const* d_in, const int* in_sizes, int n_in,
                              void* d_out, int out_size, void* d_ws, size_t ws_size,
                              hipStream_t stream) {
  const float* x     = (const float*)d_in[0];
  const int*   erow  = (const int*)d_in[1];
  const int*   ecol  = (const int*)d_in[2];
  const float* aW1   = (const float*)d_in[3];
  const float* ab1   = (const float*)d_in[4];
  const float* aW2   = (const float*)d_in[5];
  const float* ab2   = (const float*)d_in[6];
  const float* Wlin  = (const float*)d_in[7];
  const float* blin  = (const float*)d_in[8];
  const float* Wconv = (const float*)d_in[9];
  const float* bch   = (const float*)d_in[10];
  const float* Wcls  = (const float*)d_in[11];
  const float* bcls  = (const float*)d_in[12];

  int N = in_sizes[0] / NFEAT;
  int E = in_sizes[1];

  char* ws = (char*)d_ws;
  size_t off = 0;
  auto alloc = [&](size_t bytes) -> void* {
    void* p = ws + off;
    off = (off + bytes + 255) & ~(size_t)255;
    return p;
  };
  _Float16* Bswz  = (_Float16*)alloc((size_t)NFRAG * 32 * 16 * sizeof(_Float16));
  float* beff     = (float*)alloc(128 * sizeof(float));
  float* pq       = (float*)alloc((size_t)N * 8 * sizeof(float));
  float* cbuf     = (float*)alloc((size_t)N * 128 * sizeof(float));
  int* deg        = (int*)alloc((size_t)N * sizeof(int));
  int* starts     = (int*)alloc((size_t)(N + 1) * sizeof(int));
  int* cursors    = (int*)alloc((size_t)N * sizeof(int));
  int* eid        = (int*)alloc((size_t)E * sizeof(int));

  float* h      = (float*)d_out;
  float* logits = (float*)d_out + (size_t)N * 128;

  k_setup<<<1, 256, 0, stream>>>(Wlin, Wconv, aW1, blin, Bswz, beff);
  k_zero<<<(N + 255) / 256, 256, 0, stream>>>(deg, N);
  k_gemm<<<(N + 127) / 128, 256, 0, stream>>>(x, Bswz, beff, cbuf, pq, N);
  k_hist<<<(E + 255) / 256, 256, 0, stream>>>(erow, E, deg);
  k_scan<<<1, 1024, 0, stream>>>(deg, starts, cursors, N);
  k_fill<<<(E + 255) / 256, 256, 0, stream>>>(erow, E, cursors, eid);
  k_agg<<<(N + 7) / 8, 256, 0, stream>>>(cbuf, pq, starts, eid, ecol,
                                         ab1, aW2, ab2, bch, Wcls, bcls,
                                         h, logits, N);
}
